// VRNN_DAT_4655744548920
// MI455X (gfx1250) — matrix operand order repaired and checked by reading, not compiled, not run on hardware
//
#include <hip/hip_runtime.h>
#include <hip/hip_bf16.h>

typedef __attribute__((ext_vector_type(16))) __bf16 v16bf;
typedef __attribute__((ext_vector_type(8)))  float  v8f;

#define B_  256
#define T_  512
#define X_  64
#define H_  256
#define Z_  64
#define P_  32

__device__ __forceinline__ float softplus_(float x) {
    // log1p(exp(x)) numerically stable
    return fmaxf(x, 0.f) + log1pf(__expf(-fabsf(x)));
}
__device__ __forceinline__ float sigm_(float x) {
    return 1.f / (1.f + __expf(-x));
}

// ---------------------------------------------------------------------------
// Weight prep: fp32 row-major [K][Nsrc] (cols [col0,col0+Ncopy)) -> bf16,
// swizzled into WMMA B-fragment order for V_WMMA_F32_16X16X32_BF16:
// per (32K x 16N) tile, lane = (k_in>>4)*16 + n_in, element = k_in & 15.
// Each wave later loads one contiguous v16bf (32B) per lane.
// ---------------------------------------------------------------------------
__global__ void prep_swz(const float* __restrict__ src, int Nsrc, int col0,
                         __bf16* __restrict__ dst, int Ndst, int dcol0,
                         int K, int Ncopy)
{
    int idx = blockIdx.x * blockDim.x + threadIdx.x;
    if (idx >= K * Ncopy) return;
    int k = idx / Ncopy, c = idx % Ncopy;
    float v = src[(size_t)k * Nsrc + col0 + c];
    int nd   = dcol0 + c;
    int NT   = Ndst >> 4;
    int kt   = k >> 5, k_in = k & 31, nt = nd >> 4, n_in = nd & 15;
    int lane = ((k_in >> 4) << 4) | n_in;
    int elem = k_in & 15;
    dst[(((size_t)kt * NT + nt) * 32 + lane) * 16 + elem] = (__bf16)v;
}

// ---------------------------------------------------------------------------
// Cooperative GEMM: C[16,N] = act( A[16,K] @ W[K,N] + bias [+ Cold] )
// A from LDS (f32 or bf16), W pre-swizzled bf16 in global (L2-resident),
// waves of the workgroup partition the N/16 column tiles.
// Epilogue optionally writes bf16 to LDS and/or f32 to global outputs.
// act: 0 none, 1 relu, 3 split-softplus (cols >= asplit get softplus)
// ---------------------------------------------------------------------------
template <typename TIn>
__device__ __forceinline__ void gemm16(
    const TIn* __restrict__ in, int ldi, int K,
    const __bf16* __restrict__ wsw, int N,
    const float* __restrict__ bias0, const float* __restrict__ bias1, int bsplit,
    __bf16* __restrict__ out, int ldo, bool accum,
    int act, int asplit,
    float* __restrict__ g0, float* __restrict__ g1, int gsplit, long grstride)
{
    const int lane = threadIdx.x & 31;
    const int wid  = threadIdx.x >> 5;
    const int nw   = blockDim.x >> 5;
    const int m    = lane & 15;   // A row / C col within tile
    const int half = lane >> 4;
    const int NT   = N >> 4;
    const int KT   = K >> 5;

    for (int nt = wid; nt < NT; nt += nw) {
        v8f c = {};
        for (int kt = 0; kt < KT; ++kt) {
            // A fragment (16x32 bf16): lanes 0-15 hold K 0-15,
            // lanes 16-31 hold K 16-31, row = lane&15.
            v16bf a;
            const TIn* arow = in + m * ldi + kt * 32;
#pragma unroll
            for (int j = 0; j < 8; ++j) {
                int kk = half * 16 + (j << 1);
                a[2 * j]     = (__bf16)(float)arow[kk];
                a[2 * j + 1] = (__bf16)(float)arow[kk + 1];
            }
            // B fragment: one coalesced 32B load per lane
            v16bf b = *(const v16bf*)(wsw + (((size_t)kt * NT + nt) * 32 + lane) * 16);
            c = __builtin_amdgcn_wmma_f32_16x16x32_bf16(
                    false, a, false, b, (short)0, c, false, false);
        }
        const int ncol = (nt << 4) + m;
        const float bv = (ncol < bsplit) ? bias0[ncol] : bias1[ncol - bsplit];
#pragma unroll
        for (int v = 0; v < 8; ++v) {
            const int row = v + (half << 3);
            float r = c[v] + bv;
            if (accum && out) r += (float)out[row * ldo + ncol];
            if (act == 1) r = fmaxf(r, 0.f);
            else if (act == 3 && ncol >= asplit) r = softplus_(r);
            if (out) out[row * ldo + ncol] = (__bf16)r;
            if (g0) {
                if (ncol < gsplit) g0[(long)row * grstride + ncol] = r;
                else               g1[(long)row * grstride + (ncol - gsplit)] = r;
            }
        }
    }
}

struct VP {
    const float *x, *eps;
    const float *b_phi1, *b_phi2, *b_phiz;
    const float *b_enc1, *b_enc2, *b_encm, *b_encs;
    const float *b_pri, *b_prim, *b_pris;
    const float *b_dec1, *b_dec2, *b_decm, *b_decs;
    const float *gru_b;
    const __bf16 *w_phi1, *w_phi2, *w_enc1, *w_enc2, *w_encms, *w_pri, *w_prims,
                 *w_phiz, *w_dec1, *w_dec2, *w_decms, *w_gruK, *w_gruRzr, *w_gruRh;
    float *o_y, *o_phix, *o_encm, *o_encs, *o_decm, *o_decs, *o_prim, *o_pris;
};

// One workgroup per 16 batch rows; persistent over all T steps.
__global__ __launch_bounds__(512, 1) void vrnn_main(VP p)
{
    __shared__ float   h[16 * 256];     // f32 recurrent state (precision-critical)
    __shared__ __bf16  cat[16 * 320];   // concat / small-input staging
    __shared__ __bf16  t0[16 * 768];    // wide buffer (enc_h / dec_h / GRU mx)
    __shared__ __bf16  u0[16 * 256];    // ping buffer (enc1 / pri_h / dec1 / hh ...)
    __shared__ __bf16  phix[16 * 32];
    __shared__ __bf16  pz[16 * 64];

    const int tid  = threadIdx.x;
    const int nthr = blockDim.x;
    const int b0   = blockIdx.x * 16;
    const long gsZ = (long)T_ * Z_;
    const long gsP = (long)T_ * P_;
    const float* gb0 = p.gru_b;
    const float* gb1 = p.gru_b + 768;
    const int BIG = 1 << 30;

    for (int i = tid; i < 16 * 256; i += nthr) h[i] = 0.f;
    __syncthreads();

    for (int t = 0; t < T_; ++t) {
        const size_t tb = (size_t)b0 * T_ + t;

        // ---- load x_t into cat[:, 0:64]
        for (int i = tid; i < 16 * 64; i += nthr) {
            int r = i >> 6, c = i & 63;
            cat[r * 320 + c] = (__bf16)p.x[((size_t)(b0 + r) * T_ + t) * X_ + c];
        }
        __syncthreads();

        // ---- phi_x: two tiny dense+relu
        gemm16<__bf16>(cat, 320, 64,  p.w_phi1, 32, p.b_phi1, p.b_phi1, BIG,
                       u0, 256, false, 1, 0, nullptr, nullptr, 0, 0);
        __syncthreads();
        gemm16<__bf16>(u0, 256, 32,   p.w_phi2, 32, p.b_phi2, p.b_phi2, BIG,
                       phix, 32, false, 1, 0,
                       p.o_phix + tb * P_, nullptr, BIG, gsP);
        __syncthreads();

        // ---- encoder: cat=[phi_x | h]
        for (int i = tid; i < 16 * 288; i += nthr) {
            int r = i / 288, c = i % 288;
            cat[r * 320 + c] = (c < 32) ? phix[r * 32 + c]
                                        : (__bf16)h[r * 256 + (c - 32)];
        }
        __syncthreads();
        gemm16<__bf16>(cat, 320, 288, p.w_enc1, 256, p.b_enc1, p.b_enc1, BIG,
                       u0, 256, false, 1, 0, nullptr, nullptr, 0, 0);
        __syncthreads();
        gemm16<__bf16>(u0, 256, 256,  p.w_enc2, 256, p.b_enc2, p.b_enc2, BIG,
                       t0, 768, false, 1, 0, nullptr, nullptr, 0, 0);
        __syncthreads();
        // fused enc_mean|enc_std head (softplus on std half), f32 to global too
        gemm16<__bf16>(t0, 768, 256,  p.w_encms, 128, p.b_encm, p.b_encs, 64,
                       u0, 256, false, 3, 64,
                       p.o_encm + tb * Z_, p.o_encs + tb * Z_, 64, gsZ);
        __syncthreads();

        // ---- z = eps * enc_std + enc_mean ; y_pred = z ; stash bf16 z in cat[:,0:64]
        for (int i = tid; i < 16 * 64; i += nthr) {
            int r = i >> 6, c = i & 63;
            float mean = (float)u0[r * 256 + c];
            float sd   = (float)u0[r * 256 + 64 + c];
            float e    = p.eps[((size_t)(b0 + r) * T_ + t) * Z_ + c];
            float z    = e * sd + mean;
            p.o_y[((size_t)(b0 + r) * T_ + t) * Z_ + c] = z;
            cat[r * 320 + c] = (__bf16)z;
        }
        __syncthreads();

        // ---- prior head (reads f32 h directly)
        gemm16<float>(h, 256, 256,    p.w_pri, 256, p.b_pri, p.b_pri, BIG,
                      u0, 256, false, 1, 0, nullptr, nullptr, 0, 0);
        __syncthreads();
        gemm16<__bf16>(u0, 256, 256,  p.w_prims, 128, p.b_prim, p.b_pris, 64,
                       nullptr, 0, false, 3, 64,
                       p.o_prim + tb * Z_, p.o_pris + tb * Z_, 64, gsZ);
        __syncthreads();

        // ---- phi_z
        gemm16<__bf16>(cat, 320, 64,  p.w_phiz, 64, p.b_phiz, p.b_phiz, BIG,
                       pz, 64, false, 1, 0, nullptr, nullptr, 0, 0);
        __syncthreads();

        // ---- decoder: cat=[phi_z | h]
        for (int i = tid; i < 16 * 320; i += nthr) {
            int r = i / 320, c = i % 320;
            cat[r * 320 + c] = (c < 64) ? pz[r * 64 + c]
                                        : (__bf16)h[r * 256 + (c - 64)];
        }
        __syncthreads();
        gemm16<__bf16>(cat, 320, 320, p.w_dec1, 256, p.b_dec1, p.b_dec1, BIG,
                       u0, 256, false, 1, 0, nullptr, nullptr, 0, 0);
        __syncthreads();
        gemm16<__bf16>(u0, 256, 256,  p.w_dec2, 256, p.b_dec2, p.b_dec2, BIG,
                       t0, 768, false, 1, 0, nullptr, nullptr, 0, 0);
        __syncthreads();
        gemm16<__bf16>(t0, 768, 256,  p.w_decms, 128, p.b_decm, p.b_decs, 64,
                       nullptr, 0, false, 3, 64,
                       p.o_decm + tb * Z_, p.o_decs + tb * Z_, 64, gsZ);
        __syncthreads();

        // ---- GRU: catg=[phi_x | phi_z] (K=96)
        for (int i = tid; i < 16 * 96; i += nthr) {
            int r = i / 96, c = i % 96;
            cat[r * 320 + c] = (c < 32) ? phix[r * 32 + c] : pz[r * 64 + (c - 32)];
        }
        __syncthreads();
        // mx = catg @ gru_K + b0  (full 768 into t0)
        gemm16<__bf16>(cat, 320, 96,  p.w_gruK, 768, gb0, gb0, BIG,
                       t0, 768, false, 0, 0, nullptr, nullptr, 0, 0);
        __syncthreads();
        // mx[:,0:512] += h @ R[:,0:512] + b1[0:512]  (z,r gate pre-acts fused)
        gemm16<float>(h, 256, 256,    p.w_gruRzr, 512, gb1, gb1, BIG,
                      t0, 768, true, 0, 0, nullptr, nullptr, 0, 0);
        // hh = h @ R[:,512:768] + b1[512:768]
        gemm16<float>(h, 256, 256,    p.w_gruRh, 256, gb1 + 512, gb1 + 512, BIG,
                      u0, 256, false, 0, 0, nullptr, nullptr, 0, 0);
        __syncthreads();

        // ---- gates + state update (f32)
        for (int i = tid; i < 16 * 256; i += nthr) {
            int r = i >> 8, c = i & 255;
            float xz = (float)t0[r * 768 + c];          // xz+hz (+both biases)
            float xr = (float)t0[r * 768 + 256 + c];    // xr+hr
            float xh = (float)t0[r * 768 + 512 + c];    // xh (+b0h)
            float hh = (float)u0[r * 256 + c];          // hh (+b1h)
            float zg = sigm_(xz);
            float rg = sigm_(xr);
            float cand = tanhf(xh + rg * hh);
            float hv = h[r * 256 + c];
            h[r * 256 + c] = zg * hv + (1.f - zg) * cand;
        }
        __syncthreads();
    }
}

extern "C" void kernel_launch(void* const* d_in, const int* in_sizes, int n_in,
                              void* d_out, int out_size, void* d_ws, size_t ws_size,
                              hipStream_t stream) {
    (void)in_sizes; (void)n_in; (void)out_size; (void)ws_size;
    const float* x   = (const float*)d_in[0];
    const float* eps = (const float*)d_in[3];

    // ---- bf16 swizzled weight arena in d_ws
    __bf16* w = (__bf16*)d_ws;
    size_t off = 0;
    auto alloc = [&](size_t n) { __bf16* q = w + off; off += n; return q; };
    __bf16* w_phi1   = alloc(64  * 32);
    __bf16* w_phi2   = alloc(32  * 32);
    __bf16* w_enc1   = alloc(288 * 256);
    __bf16* w_enc2   = alloc(256 * 256);
    __bf16* w_encms  = alloc(256 * 128);
    __bf16* w_pri    = alloc(256 * 256);
    __bf16* w_prims  = alloc(256 * 128);
    __bf16* w_phiz   = alloc(64  * 64);
    __bf16* w_dec1   = alloc(320 * 256);
    __bf16* w_dec2   = alloc(256 * 256);
    __bf16* w_decms  = alloc(256 * 128);
    __bf16* w_gruK   = alloc(96  * 768);
    __bf16* w_gruRzr = alloc(256 * 512);
    __bf16* w_gruRh  = alloc(256 * 256);

    auto prep = [&](int src_idx, int Nsrc, int col0,
                    __bf16* dst, int Ndst, int dcol0, int K, int Ncopy) {
        int n = K * Ncopy;
        prep_swz<<<(n + 255) / 256, 256, 0, stream>>>(
            (const float*)d_in[src_idx], Nsrc, col0, dst, Ndst, dcol0, K, Ncopy);
    };
    prep(4,  32, 0,   w_phi1,   32, 0,  64, 32);    // phi_x_W1
    prep(6,  32, 0,   w_phi2,   32, 0,  32, 32);    // phi_x_W2
    prep(10, 256, 0,  w_enc1,  256, 0, 288, 256);   // enc_W1
    prep(12, 256, 0,  w_enc2,  256, 0, 256, 256);   // enc_W2
    prep(14, 64, 0,   w_encms, 128, 0,  256, 64);   // enc_mean_W -> cols 0:64
    prep(16, 64, 0,   w_encms, 128, 64, 256, 64);   // enc_std_W  -> cols 64:128
    prep(18, 256, 0,  w_pri,   256, 0, 256, 256);   // prior_W
    prep(20, 64, 0,   w_prims, 128, 0,  256, 64);   // prior_mean_W
    prep(22, 64, 0,   w_prims, 128, 64, 256, 64);   // prior_std_W
    prep(8,  64, 0,   w_phiz,   64, 0,  64, 64);    // phi_z_W
    prep(24, 256, 0,  w_dec1,  256, 0, 320, 256);   // dec_W1
    prep(26, 256, 0,  w_dec2,  256, 0, 256, 256);   // dec_W2
    prep(28, 64, 0,   w_decms, 128, 0,  256, 64);   // dec_mean_W
    prep(30, 64, 0,   w_decms, 128, 64, 256, 64);   // dec_std_W
    prep(32, 768, 0,  w_gruK,  768, 0,  96, 768);   // gru_K
    prep(33, 768, 0,  w_gruRzr, 512, 0, 256, 512);  // gru_R[:, 0:512]
    prep(33, 768, 512, w_gruRh, 256, 0, 256, 256);  // gru_R[:, 512:768]

    // ---- output slices (flat, reference return order)
    float* out = (float*)d_out;
    const size_t szY  = (size_t)B_ * T_ * Z_;
    const size_t szPX = (size_t)B_ * T_ * P_;
    VP p{};
    p.x = x; p.eps = eps;
    p.b_phi1 = (const float*)d_in[5];
    p.b_phi2 = (const float*)d_in[7];
    p.b_phiz = (const float*)d_in[9];
    p.b_enc1 = (const float*)d_in[11];
    p.b_enc2 = (const float*)d_in[13];
    p.b_encm = (const float*)d_in[15];
    p.b_encs = (const float*)d_in[17];
    p.b_pri  = (const float*)d_in[19];
    p.b_prim = (const float*)d_in[21];
    p.b_pris = (const float*)d_in[23];
    p.b_dec1 = (const float*)d_in[25];
    p.b_dec2 = (const float*)d_in[27];
    p.b_decm = (const float*)d_in[29];
    p.b_decs = (const float*)d_in[31];
    p.gru_b  = (const float*)d_in[34];
    p.w_phi1 = w_phi1;  p.w_phi2 = w_phi2;  p.w_enc1 = w_enc1;  p.w_enc2 = w_enc2;
    p.w_encms = w_encms; p.w_pri = w_pri;   p.w_prims = w_prims; p.w_phiz = w_phiz;
    p.w_dec1 = w_dec1;  p.w_dec2 = w_dec2;  p.w_decms = w_decms;
    p.w_gruK = w_gruK;  p.w_gruRzr = w_gruRzr; p.w_gruRh = w_gruRh;
    p.o_y    = out;
    p.o_phix = out + szY;
    p.o_encm = out + szY + szPX;
    p.o_encs = p.o_encm + szY;
    p.o_decm = p.o_encs + szY;
    p.o_decs = p.o_decm + szY;
    p.o_prim = p.o_decs + szY;
    p.o_pris = p.o_prim + szY;

    vrnn_main<<<B_ / 16, 512, 0, stream>>>(p);
}
